// d2sdm_57904749084789
// MI455X (gfx1250) — matrix operand order repaired and checked by reading, not compiled, not run on hardware
//
#include <hip/hip_runtime.h>
#include <hip/hip_bf16.h>

// ---------------------------------------------------------------- types
typedef __attribute__((ext_vector_type(16))) _Float16 v16h;
typedef __attribute__((ext_vector_type(8)))  float    v8f;
typedef __attribute__((ext_vector_type(4)))  unsigned int v4u;
typedef __attribute__((ext_vector_type(8)))  int      v8i;
typedef __attribute__((ext_vector_type(4)))  int      v4i;

union H16 { v16h v; uint4 q[2]; };
union H8  { _Float16 h[8]; uint4 q; };
union H4  { _Float16 h[4]; uint2 q; };

// ---------------------------------------------------------------- config
#define B_N    4096
#define M_N    16384
#define D_K    512
#define NCP    128          // 100 classes + ones-column (c=100) + zero pad
#define BT     128          // queries per workgroup
#define MT     64           // memory rows per inner tile
#define MSPLIT 8
#define MPW    (M_N / MSPLIT)   // 2048 rows per WG
#define ITERS  (MPW / MT)       // 32
#define XS_LD  520          // 512 + 8 halfs pad -> conflict-free ds_load_b128
#define AS_LD  72           // 64 + 8 halfs pad

__device__ __forceinline__ v8f wmma_f16(v16h a, v16h b, v8f c) {
  return __builtin_amdgcn_wmma_f32_16x16x32_f16(false, a, false, b, (short)0, c,
                                                false, false);
}

// A-operand fragment: 16-bit A 16x32 layout. lane<16: K {k0..k0+7},{k0+16..23};
// lane>=16: +8. Source is row-major [row][k].
__device__ __forceinline__ v16h frag_a(const _Float16* mem, int rowBase, int k0,
                                       int stride, int lane) {
  int r = rowBase + (lane & 15);
  int off = (lane >> 4) << 3;
  const _Float16* p = mem + r * stride + k0 + off;
  H16 u;
  u.q[0] = *(const uint4*)(p);
  u.q[1] = *(const uint4*)(p + 16);
  return u.v;
}

// B-operand fragment: lane holds column (lane&15), K {k0..k0+7},{k0+16..23};
// lane>=16: +8. Source is row-major B^T, i.e. [n][k].
__device__ __forceinline__ v16h frag_b(const _Float16* mem, int rowBase, int k0,
                                       int stride, int lane) {
  int r = rowBase + (lane & 15);
  int off = (lane >> 4) << 3;
  const _Float16* p = mem + r * stride + k0 + off;
  H16 u;
  u.q[0] = *(const uint4*)(p);
  u.q[1] = *(const uint4*)(p + 16);
  return u.v;
}

// Tensor Data Mover: async DMA of a 2-D f16 tile (tile1 rows x tile0 cols,
// row stride `stride` elements) from global into LDS, adding `pad_amt`
// (code: (n+1) DWORDs) every `pad_iv` (code: 2^(n+1) DWORDs) for bank-conflict
// free fragment loads.  D# packing per CDNA5 ISA 8.3/8.4.  Tracked by TENSORcnt.
// This toolchain exposes the 6-arg builtin; groups 2/3 unused for 2-D tiles.
__device__ __forceinline__ void tdm_load_2d(unsigned lds_addr, const void* gptr,
                                            unsigned dim0, unsigned dim1,
                                            unsigned tile0, unsigned tile1,
                                            unsigned stride, unsigned pad_iv,
                                            unsigned pad_amt) {
  unsigned long long ga = (unsigned long long)(uintptr_t)gptr;
  v4u g0;
  g0[0] = 1u;                                   // count=1, user D#
  g0[1] = lds_addr;                             // LDS byte address
  g0[2] = (unsigned)ga;                         // global_addr[31:0]
  g0[3] = (unsigned)(ga >> 32) | (2u << 30);    // global_addr[56:32] | type=2
  v8i g1;
  g1[0] = (int)((1u << 16) |                    // data_size = 2 bytes
                (1u << 20) |                    // pad_enable
                (pad_iv << 22) | (pad_amt << 25));
  g1[1] = (int)(dim0 << 16);                    // tensor_dim0[15:0]
  g1[2] = (int)((dim0 >> 16) | (dim1 << 16));   // dim0[31:16] | dim1[15:0]
  g1[3] = (int)((dim1 >> 16) | (tile0 << 16));  // dim1[31:16] | tile_dim0
  g1[4] = (int)tile1;                           // tile_dim1 (tile_dim2=0 -> 2D)
  g1[5] = (int)stride;                          // tensor_dim0_stride[31:0]
  g1[6] = 0;
  g1[7] = 0;
  v4i z4 = (v4i){0, 0, 0, 0};
  v8i z8 = (v8i){0, 0, 0, 0, 0, 0, 0, 0};
  __builtin_amdgcn_tensor_load_to_lds(g0, g1, z4, z4, z8, 0);
}

// ---------------------------------------------------------------- prepass
// f32 row-major (rows x 512) -> f16 copy + sum of squares of the f16 values.
__global__ __launch_bounds__(256) void d2sdm_conv_rows(
    const float* __restrict__ src, _Float16* __restrict__ dst,
    float* __restrict__ sq) {
  int row  = blockIdx.x * 8 + (threadIdx.x >> 5);
  int lane = threadIdx.x & 31;
  const float* s = src + (size_t)row * D_K;
  _Float16*    d = dst + (size_t)row * D_K;
  float acc = 0.f;
  for (int k = 0; k < 4; ++k) {
    int idx = (k * 32 + lane) * 4;
    float4 v = *(const float4*)(s + idx);
    H4 u;
    u.h[0] = (_Float16)v.x; u.h[1] = (_Float16)v.y;
    u.h[2] = (_Float16)v.z; u.h[3] = (_Float16)v.w;
    *(uint2*)(d + idx) = u.q;
    acc += (float)u.h[0] * (float)u.h[0] + (float)u.h[1] * (float)u.h[1] +
           (float)u.h[2] * (float)u.h[2] + (float)u.h[3] * (float)u.h[3];
  }
  for (int o = 16; o > 0; o >>= 1) acc += __shfl_xor(acc, o, 32);
  if (lane == 0) sq[row] = acc;
}

// C (M x 100 f32) -> Ct (NCP x M f16), c==100 row = ones (free normalizer).
__global__ __launch_bounds__(256) void d2sdm_conv_c(
    const float* __restrict__ C, _Float16* __restrict__ Ct) {
  int m = blockIdx.x * 256 + threadIdx.x;
  for (int c = 0; c < NCP; ++c) {
    float v = (c < 100) ? C[(size_t)m * 100 + c] : (c == 100 ? 1.0f : 0.0f);
    Ct[(size_t)c * M_N + m] = (_Float16)v;
  }
}

// ---------------------------------------------------------------- main
__global__ __launch_bounds__(256) void d2sdm_fused(
    const _Float16* __restrict__ Ah, const _Float16* __restrict__ Xh,
    const _Float16* __restrict__ Ct, const float* __restrict__ aaG,
    const float* __restrict__ xxG, float* __restrict__ pAcc) {
  __shared__ _Float16 Xs[BT * XS_LD];        // 133,120 B resident X block
  __shared__ _Float16 AWs[2 * MT * AS_LD];   // 18,432 B  A double-buffer == Ws
  __shared__ float aas[MT];
  __shared__ float xxs[BT];

  const int tid  = threadIdx.x;
  const int lane = tid & 31;
  const int wv   = tid >> 5;                 // 8 waves
  const int b0   = blockIdx.x * BT;
  const int mbase = blockIdx.y * MPW;

  // One TDM op stages the whole resident X block (128 x 512 f16, +16B row pad).
  if (wv == 0) {
    tdm_load_2d((unsigned)(uintptr_t)&Xs[0], &Xh[(size_t)b0 * D_K],
                D_K, BT, D_K, BT, D_K, /*pad_iv 256dw=1024B*/ 7,
                /*pad_amt 4dw=16B*/ 3);
  }
  if (tid < BT) xxs[tid] = xxG[b0 + tid];

  const int mg  = wv & 1;              // m-tile pair: tiles 2mg, 2mg+1
  const int bgp = wv >> 1;             // b-tile pair: tiles 2bgp, 2bgp+1
  const int hi8 = (lane >> 4) << 3;

  v8f acc2[8];                         // (b-tile 0..7) x (c-tile = wv)
  for (int t = 0; t < 8; ++t) acc2[t] = (v8f){0, 0, 0, 0, 0, 0, 0, 0};

  for (int iter = 0; iter < ITERS; ++iter) {
    const int mt = mbase + iter * MT;
    __syncthreads();                   // prev GEMM2 done -> AWs region free

    if (tid < MT) aas[tid] = aaG[mt + tid];
    if (wv == 0) {                     // stage chunk 0 -> buffer 0 (async)
      tdm_load_2d((unsigned)(uintptr_t)&AWs[0], &Ah[(size_t)mt * D_K],
                  64, MT, 64, MT, D_K, /*pad_iv 32dw=128B*/ 4,
                  /*pad_amt 4dw=16B*/ 3);
    }

    v8f acc1[4];                       // [mi][bj] 2x2
    for (int t = 0; t < 4; ++t) acc1[t] = (v8f){0, 0, 0, 0, 0, 0, 0, 0};

    for (int kc = 0; kc < 8; ++kc) {
      if (wv == 0) __builtin_amdgcn_s_wait_tensorcnt(0);  // chunk kc resident
      __syncthreads();                 // all waves: chunk kc visible, kc-1 done
      if (wv == 0 && kc < 7) {         // async DMA of next chunk over compute
        tdm_load_2d((unsigned)(uintptr_t)&AWs[((kc + 1) & 1) * (MT * AS_LD)],
                    &Ah[(size_t)mt * D_K + (kc + 1) * 64],
                    64, MT, 64, MT, D_K, 4, 3);
      }
      const _Float16* As = AWs + (kc & 1) * (MT * AS_LD);
      for (int ks = 0; ks < 2; ++ks) {
        int k0 = ks * 32;
        int kg = kc * 64 + k0;
        v16h af0 = frag_a(As, (mg * 2 + 0) * 16, k0, AS_LD, lane);
        v16h af1 = frag_a(As, (mg * 2 + 1) * 16, k0, AS_LD, lane);
        v16h xf0 = frag_b(Xs, (bgp * 2 + 0) * 16, kg, XS_LD, lane);
        v16h xf1 = frag_b(Xs, (bgp * 2 + 1) * 16, kg, XS_LD, lane);
        acc1[0] = wmma_f16(af0, xf0, acc1[0]);
        acc1[1] = wmma_f16(af0, xf1, acc1[1]);
        acc1[2] = wmma_f16(af1, xf0, acc1[2]);
        acc1[3] = wmma_f16(af1, xf1, acc1[3]);
      }
    }

    __syncthreads();                   // all A reads done; reuse region as Ws
    _Float16* Ws = AWs;                // [BT][AS_LD] : w^T, row-major [b][m]
    for (int mi = 0; mi < 2; ++mi) {
      int mtile = mg * 2 + mi;
      const float* aap = &aas[mtile * 16 + hi8];
      for (int bj = 0; bj < 2; ++bj) {
        int bl = (bgp * 2 + bj) * 16 + (lane & 15);
        float xxv = xxs[bl];
        v8f d = acc1[mi * 2 + bj];
        H8 wu;
        for (int r = 0; r < 8; ++r) {
          float d2 = aap[r] + xxv - 2.0f * d[r];    // ||a-x||^2 (f16-consistent)
          float dist = __builtin_sqrtf(fmaxf(d2, 0.0f));
          // exp(-dist/4) = exp2(dist * -log2(e)/4); range [e^-10, 1]: safe.
          wu.h[r] = (_Float16)__builtin_exp2f(dist * -0.3606737602222409f);
        }
        *(uint4*)&Ws[bl * AS_LD + mtile * 16 + hi8] = wu.q;
      }
    }

    __syncthreads();                   // Ws visible to all waves
    for (int ks = 0; ks < 2; ++ks) {   // GEMM2: acc2 += Ws(b x m) * C(m x c)
      int k0 = ks * 32;
      v16h cf = frag_b(Ct, wv * 16, mt + k0, M_N, lane);  // C^T rows (L2-hot)
      for (int bt = 0; bt < 8; ++bt) {
        v16h wf = frag_a(Ws, bt * 16, k0, AS_LD, lane);
        acc2[bt] = wmma_f16(wf, cf, acc2[bt]);
      }
    }
  }

  // Write deterministic partials: pAcc[split][b][c]
  float* dst = pAcc + ((size_t)blockIdx.y * B_N + b0) * NCP;
  int c = wv * 16 + (lane & 15);
  for (int bt = 0; bt < 8; ++bt) {
    int bl = bt * 16 + hi8;
    for (int r = 0; r < 8; ++r)
      dst[(size_t)(bl + r) * NCP + c] = acc2[bt][r];
  }
}

// ---------------------------------------------------------------- reduce
__global__ __launch_bounds__(128) void d2sdm_reduce(
    const float* __restrict__ pAcc, float* __restrict__ out) {
  int b = blockIdx.x, c = threadIdx.x;
  float asum = 0.f, lsum = 0.f;
  for (int s = 0; s < MSPLIT; ++s) {
    const float* p = pAcc + ((size_t)s * B_N + b) * NCP;
    asum += p[c];
    lsum += p[100];                    // ones-column = softmin denominator
  }
  if (c < 100) out[(size_t)b * 100 + c] = asum / lsum;
}

// ---------------------------------------------------------------- launch
extern "C" void kernel_launch(void* const* d_in, const int* in_sizes, int n_in,
                              void* d_out, int out_size, void* d_ws,
                              size_t ws_size, hipStream_t stream) {
  const float* X = (const float*)d_in[0];   // (4096, 512)
  const float* A = (const float*)d_in[1];   // (16384, 512)
  const float* C = (const float*)d_in[2];   // (16384, 100)
  float* out = (float*)d_out;               // (4096, 100)

  char* w = (char*)d_ws;
  _Float16* Ah = (_Float16*)w; w += (size_t)M_N * D_K * 2;   // 16.78 MB
  _Float16* Xh = (_Float16*)w; w += (size_t)B_N * D_K * 2;   //  4.19 MB
  _Float16* Ct = (_Float16*)w; w += (size_t)NCP * M_N * 2;   //  4.19 MB
  float* aa   = (float*)w;     w += (size_t)M_N * 4;         //  64 KB
  float* xx   = (float*)w;     w += (size_t)B_N * 4;         //  16 KB
  float* pAcc = (float*)w;     w += (size_t)MSPLIT * B_N * NCP * 4; // 16.78 MB

  d2sdm_conv_rows<<<M_N / 8, 256, 0, stream>>>(A, Ah, aa);
  d2sdm_conv_rows<<<B_N / 8, 256, 0, stream>>>(X, Xh, xx);
  d2sdm_conv_c<<<M_N / 256, 256, 0, stream>>>(C, Ct);
  d2sdm_fused<<<dim3(B_N / BT, MSPLIT), 256, 0, stream>>>(Ah, Xh, Ct, aa, xx,
                                                          pAcc);
  d2sdm_reduce<<<B_N, 128, 0, stream>>>(pAcc, out);
}
